// CHConv_69174743269742
// MI455X (gfx1250) — compile-verified, matrix operand order checked by reading, not run on hardware
//
#include <hip/hip_runtime.h>

// CDNA5 / gfx1250: wave32, WMMA 16x16x32 bf16 -> f32
typedef __attribute__((ext_vector_type(16))) __bf16       v16bf;
typedef __attribute__((ext_vector_type(8)))  float        v8f;
typedef __attribute__((ext_vector_type(8)))  unsigned int v8u;

// Problem constants (from reference): B=2,H=128,W=256,C=64,F=128,kH=kW=3
#define CH_B      2
#define CH_H      128
#define CH_W      256
#define CH_C      64
#define CH_F      128
#define CH_K      9
#define CH_CK     576           // C*K, flattened as ck = tap*64 + c
#define CH_CHUNKS 18            // CK / 32
#define M_TILE    32            // output positions per block (2 WMMA M tiles)
#define N_PAIRS   288           // M_TILE * 9 taps
#define N_SLOTS   2304          // N_PAIRS * 8 channel-groups (8 ch each)

static __device__ __forceinline__ unsigned short f32_to_bf16_rne(float f) {
    unsigned int b = __builtin_bit_cast(unsigned int, f);
    return (unsigned short)((b + 0x7fffu + ((b >> 16) & 1u)) >> 16);
}

// ---------------------------------------------------------------------------
// Prepass: repack kernel[F][C][3][3] (f32) into bf16 B-matrix fragments.
// Layout: Bpacked[(chunk*2 + half)*128 + n][16], 16 bf16 entries per lane in
// the 16-bit WMMA fragment K-pair pattern: kk = (idx&7) + 8*half + 16*(idx>>3)
// ck = chunk*32 + kk, with ck = tap*64 + c  ->  kernel[n][c][tap].
// ---------------------------------------------------------------------------
__global__ void chconv_pack_b_bf16(const float* __restrict__ kern,
                                   unsigned short* __restrict__ bpk) {
    int t = blockIdx.x * blockDim.x + threadIdx.x;   // 0 .. 18*2*128*16 - 1
    if (t >= CH_CHUNKS * 2 * CH_F * 16) return;
    int idx   = t & 15;
    int n     = (t >> 4) & 127;
    int half  = (t >> 11) & 1;
    int chunk = t >> 12;
    int kk = (idx & 7) + (half << 3) + ((idx >> 3) << 4);
    int ck = chunk * 32 + kk;            // tap*64 + c
    int tap = ck >> 6;
    int c   = ck & 63;
    bpk[t] = f32_to_bf16_rne(kern[(n * CH_C + c) * CH_K + tap]);
}

// ---------------------------------------------------------------------------
// Main kernel: one block = 32 output positions (2 M tiles) x all F=128.
//  1a: pairs (pos,tap) compute bilinear params once -> LDS (2 loops)
//  1b: 2304 slots (pair x 8-channel group): gather + blend -> swizzled A LDS
//  2 : per wave, per chunk: 1x B fragment load feeds 2x wmma (A0, A1)
// ---------------------------------------------------------------------------
__global__ __launch_bounds__(256)
void chconv_wmma(const float* __restrict__ x,
                 const float* __restrict__ scale,
                 const float* __restrict__ offb,
                 const unsigned short* __restrict__ bpk,
                 float* __restrict__ out) {
    // A tiles: 2 (M tiles) * 18 chunks * 32 lanes * 16 bf16 = 36 KB
    __shared__ __align__(16) unsigned short Alds[2 * CH_CHUNKS * 32 * 16];
    // Per-pair params: 4 corner base offsets, 4 masked weights, m, tap*64
    // stride 12 dwords (48 B) keeps both uint4 groups 16-B aligned.
    __shared__ __align__(16) unsigned int Pld[N_PAIRS * 12];

    const int p0 = blockIdx.x * M_TILE;  // first of 32 flattened bhw positions
    const int t  = threadIdx.x;

    // -------- Phase 1a: bilinear parameters, computed once per pair --------
    #pragma unroll
    for (int i = 0; i < 2; ++i) {
        int pair = t + i * 256;
        if (pair < N_PAIRS) {
            int m   = pair / CH_K;
            int tap = pair % CH_K;
            int pos = p0 + m;
            int b   = pos >> 15;             // / (H*W) = 32768
            int rem = pos & 32767;
            int h   = rem >> 8;              // / W = 256
            int w   = rem & 255;

            int o18 = (h * CH_W + w) * 18 + tap * 2;
            float offy = offb[o18]     * scale[o18];
            float offx = offb[o18 + 1] * scale[o18 + 1];

            float ys = (float)(h - 1 + tap / 3) + offy;   // pad_h=1, ti=tap/3
            float xs = (float)(w - 1 + tap % 3) + offx;   // pad_w=1, tj=tap%3
            float y0f = floorf(ys), x0f = floorf(xs);
            float fy = ys - y0f,    fx = xs - x0f;
            int iy0 = (int)y0f,     ix0 = (int)x0f;

            float wg[4] = { (1.f - fy) * (1.f - fx), (1.f - fy) * fx,
                            fy * (1.f - fx),         fy * fx };

            unsigned int* P = &Pld[pair * 12];
            #pragma unroll
            for (int corner = 0; corner < 4; ++corner) {
                int dy = corner >> 1, dx = corner & 1;
                int yi = iy0 + dy,    xi = ix0 + dx;
                bool valid = (yi >= 0) && (yi < CH_H) && (xi >= 0) && (xi < CH_W);
                int yc = min(max(yi, 0), CH_H - 1);
                int xc = min(max(xi, 0), CH_W - 1);
                P[corner]     = (unsigned int)(((((b << 7) + yc) << 8) + xc) << 6);
                P[4 + corner] = __builtin_bit_cast(unsigned int,
                                                   valid ? wg[corner] : 0.0f);
            }
            P[8] = (unsigned int)m;
            P[9] = (unsigned int)(tap << 6);   // tap*64 = ck base
        }
    }
    __syncthreads();

    // -------- Phase 1b: gather + blend, 8 channels per slot ----------------
    // Exactly 9 full iterations: 2304 slots / 256 threads, no bounds check.
    #pragma unroll
    for (int i = 0; i < 9; ++i) {
        int slot = t + i * 256;
        int pair = slot >> 3;
        int c0   = (slot & 7) << 3;          // channel group of 8

        const unsigned int* P = &Pld[pair * 12];
        uint4  ad = *reinterpret_cast<const uint4*>(P);
        float4 wt = *reinterpret_cast<const float4*>(P + 4);
        int m     = (int)P[8];
        int tap64 = (int)P[9];

        float a[8] = {0.f, 0.f, 0.f, 0.f, 0.f, 0.f, 0.f, 0.f};
        unsigned int adv[4] = {ad.x, ad.y, ad.z, ad.w};
        float        wtv[4] = {wt.x, wt.y, wt.z, wt.w};
        #pragma unroll
        for (int corner = 0; corner < 4; ++corner) {
            const float4* src = reinterpret_cast<const float4*>(
                x + adv[corner] + c0);
            float4 v0 = src[0];
            float4 v1 = src[1];
            float wv = wtv[corner];
            a[0] += wv * v0.x; a[1] += wv * v0.y;
            a[2] += wv * v0.z; a[3] += wv * v0.w;
            a[4] += wv * v1.x; a[5] += wv * v1.y;
            a[6] += wv * v1.z; a[7] += wv * v1.w;
        }

        // ck base is 8-aligned -> all 8 bf16 land contiguously in one
        // fragment half: single 16-byte LDS store.
        int ck0   = tap64 + c0;
        int chunk = ck0 >> 5;
        int kk0   = ck0 & 31;                 // multiple of 8
        int half  = (kk0 >> 3) & 1;
        int mtile = m >> 4;
        int lane  = (m & 15) + (half << 4);
        int idx0  = (kk0 >> 4) << 3;          // 0 or 8
        uint4 pk;
        pk.x = (unsigned int)f32_to_bf16_rne(a[0]) |
               ((unsigned int)f32_to_bf16_rne(a[1]) << 16);
        pk.y = (unsigned int)f32_to_bf16_rne(a[2]) |
               ((unsigned int)f32_to_bf16_rne(a[3]) << 16);
        pk.z = (unsigned int)f32_to_bf16_rne(a[4]) |
               ((unsigned int)f32_to_bf16_rne(a[5]) << 16);
        pk.w = (unsigned int)f32_to_bf16_rne(a[6]) |
               ((unsigned int)f32_to_bf16_rne(a[7]) << 16);
        *reinterpret_cast<uint4*>(
            &Alds[((mtile * CH_CHUNKS + chunk) * 32 + lane) * 16 + idx0]) = pk;
    }
    __syncthreads();

    // -------- Phase 2: WMMA GEMM. One B fragment feeds two M tiles. --------
    const int lane = t & 31;
    const int wave = t >> 5;
    const int n    = (wave << 4) + (lane & 15);  // output feature column
    const int hsel = lane >> 4;                  // fragment half select

    const v8u* ap = reinterpret_cast<const v8u*>(Alds);
    const v8u* bp = reinterpret_cast<const v8u*>(bpk);

    v8f acc0 = {};
    v8f acc1 = {};
    for (int c = 0; c < CH_CHUNKS; ++c) {
        v16bf bm = __builtin_bit_cast(v16bf, bp[(c * 2 + hsel) * CH_F + n]); // 2x global b128
        v16bf a0 = __builtin_bit_cast(v16bf, ap[c * 32 + lane]);             // 2x ds_load_b128
        v16bf a1 = __builtin_bit_cast(v16bf, ap[(CH_CHUNKS + c) * 32 + lane]);
        acc0 = __builtin_amdgcn_wmma_f32_16x16x32_bf16(
            false, a0, false, bm, (short)0, acc0, false, false);
        acc1 = __builtin_amdgcn_wmma_f32_16x16x32_bf16(
            false, a1, false, bm, (short)0, acc1, false, false);
    }

    // D layout: lanes 0-15 hold M=r, lanes 16-31 hold M=r+8 (VGPR r = 0..7)
    #pragma unroll
    for (int r = 0; r < 8; ++r) {
        int m = r + (hsel << 3);
        out[(p0 + m) * CH_F + n]      = acc0[r];
        out[(p0 + 16 + m) * CH_F + n] = acc1[r];
    }
}

// ---------------------------------------------------------------------------
extern "C" void kernel_launch(void* const* d_in, const int* in_sizes, int n_in,
                              void* d_out, int out_size, void* d_ws, size_t ws_size,
                              hipStream_t stream) {
    const float* x     = (const float*)d_in[0];   // [2,128,256,64]
    const float* kern  = (const float*)d_in[1];   // [128,64,3,3]
    const float* scale = (const float*)d_in[2];   // [128,256,18]
    const float* offb  = (const float*)d_in[3];   // [128,256,18]
    float*       out   = (float*)d_out;           // [2,128,256,128]

    unsigned short* bpk = (unsigned short*)d_ws;  // 18*2*128*16 bf16 = 144 KB

    chconv_pack_b_bf16<<<(CH_CHUNKS * 2 * CH_F * 16 + 255) / 256, 256, 0, stream>>>(kern, bpk);
    chconv_wmma<<<(CH_B * CH_H * CH_W) / M_TILE, 256, 0, stream>>>(x, scale, offb, bpk, out);
}
